// GraphX_86973087744606
// MI455X (gfx1250) — compile-verified, matrix operand order checked by reading, not run on hardware
//
#include <hip/hip_runtime.h>
#include <hip/hip_bf16.h>
#include <math.h>

// ---------------------------------------------------------------------------
// GraphX layer for MI455X (gfx1250, wave32).
// GEMMs (~25 GF total) run on v_wmma_f32_16x16x32_bf16 (AI ~64 f/B => compute
// bound; bf16 matrix path is the chip's fast dense pipe, f32 accumulate keeps
// precision). Edge phase (softmax-aggregation) is memory bound; Q/K/V (25.6MB
// each) are L2-resident (192MB L2) so per-edge gathers + f32 atomics are the
// right path.
// ---------------------------------------------------------------------------

#define NSRC   50000
#define NDST   50000
#define NEDGE  800000
#define DM     128
#define NH     8
#define DHD    16
#define MAXBUCK 512

typedef __attribute__((ext_vector_type(8)))  __bf16 v8bf;
typedef __attribute__((ext_vector_type(16))) __bf16 v16bf;
typedef __attribute__((ext_vector_type(8)))  float  v8f;

__device__ __forceinline__ __bf16 f2bf(float f) {
  unsigned u = __builtin_bit_cast(unsigned, f);
  unsigned r = u + 0x7FFFu + ((u >> 16) & 1u);   // round-to-nearest-even
  unsigned short h = (unsigned short)(r >> 16);
  return __builtin_bit_cast(__bf16, h);
}
__device__ __forceinline__ float sigm(float x) { return 1.0f / (1.0f + __expf(-x)); }

// Monotone float<->uint mapping so atomicMax(uint) orders like float max.
#define SEGMAX_INIT 0x007FFFFFu   // == fkey(-inf)
__device__ __forceinline__ unsigned fkey(float x) {
  unsigned u = __builtin_bit_cast(unsigned, x);
  return u ^ ((u & 0x80000000u) ? 0xFFFFFFFFu : 0x80000000u);
}
__device__ __forceinline__ float funkey(unsigned u) {
  if (u == SEGMAX_INIT) return 0.0f;            // empty segment -> where(isfinite,...,0)
  unsigned b = (u & 0x80000000u) ? (u ^ 0x80000000u) : ~u;
  return __builtin_bit_cast(float, b);
}

__device__ __forceinline__ float wave_sum32(float s) {
  s += __shfl_xor(s, 1, 32);
  s += __shfl_xor(s, 2, 32);
  s += __shfl_xor(s, 4, 32);
  s += __shfl_xor(s, 8, 32);
  s += __shfl_xor(s, 16, 32);
  return s;
}

// ---------------------------------------------------------------------------
// Degrees
// ---------------------------------------------------------------------------
__global__ void deg_kernel(const int* __restrict__ eu, const int* __restrict__ ev,
                           int* __restrict__ deg_out, int* __restrict__ deg_in) {
  int e = blockIdx.x * blockDim.x + threadIdx.x;
  if (e >= NEDGE) return;
  atomicAdd(&deg_out[eu[e]], 1);
  atomicAdd(&deg_in[ev[e]], 1);
}

__global__ void segmax_init_kernel(unsigned* __restrict__ seg, int n) {
  int i = blockIdx.x * blockDim.x + threadIdx.x;
  if (i < n) seg[i] = SEGMAX_INIT;
}

// ---------------------------------------------------------------------------
// fp32 -> bf16 converts (weights + activations)
// ---------------------------------------------------------------------------
__global__ void f2bf_kernel(const float* __restrict__ in, __bf16* __restrict__ out, int n) {
  int i = blockIdx.x * blockDim.x + threadIdx.x;
  if (i < n) out[i] = f2bf(in[i]);
}
// in[K,N] row-major  ->  out[N,K] row-major (for X @ W, B operand wants K rows)
__global__ void f2bfT_kernel(const float* __restrict__ in, __bf16* __restrict__ out,
                             int K, int N) {
  int i = blockIdx.x * blockDim.x + threadIdx.x;
  if (i >= K * N) return;
  int n = i / K, k = i - n * K;
  out[i] = f2bf(in[(size_t)k * N + n]);
}

// ---------------------------------------------------------------------------
// Fused (optional centrality) + LayerNorm, one wave32 per row of 128.
// deg==nullptr: plain LN. raw_out!=nullptr: also emit pre-LN row as bf16.
// ---------------------------------------------------------------------------
__global__ void cent_ln_kernel(const float* __restrict__ h,
                               const int* __restrict__ deg, int deg_is_out,
                               const float* __restrict__ emb_in,
                               const float* __restrict__ emb_out,
                               const float* __restrict__ alpha_p,
                               const float* __restrict__ lng,
                               const float* __restrict__ lnb,
                               __bf16* __restrict__ ln_out,
                               __bf16* __restrict__ raw_out,
                               int rows) {
  int row = blockIdx.x * (blockDim.x >> 5) + (threadIdx.x >> 5);
  if (row >= rows) return;
  int lane = threadIdx.x & 31;
  int d0 = lane * 4;
  const float* hr = h + (size_t)row * DM + d0;
  float x[4];
  if (deg) {
    float dv = (float)deg[row];
    int bb = (int)floorf(log1pf(dv) * (1.0f / logf(1.6f)));
    bb = bb < 0 ? 0 : (bb > MAXBUCK ? MAXBUCK : bb);
    int bi = deg_is_out ? 0 : bb;
    int bo = deg_is_out ? bb : 0;
    float a = alpha_p[0];
    const float* ei = emb_in + (size_t)bi * DM + d0;
    const float* eo = emb_out + (size_t)bo * DM + d0;
#pragma unroll
    for (int i = 0; i < 4; i++) x[i] = hr[i] + a * (ei[i] + eo[i]);
  } else {
#pragma unroll
    for (int i = 0; i < 4; i++) x[i] = hr[i];
  }
  if (raw_out) {
    __bf16* ro = raw_out + (size_t)row * DM + d0;
#pragma unroll
    for (int i = 0; i < 4; i++) ro[i] = f2bf(x[i]);
  }
  float mu = wave_sum32(x[0] + x[1] + x[2] + x[3]) * (1.0f / DM);
  float vs = 0.0f;
#pragma unroll
  for (int i = 0; i < 4; i++) { float t = x[i] - mu; vs += t * t; }
  float var = wave_sum32(vs) * (1.0f / DM);
  float rs = rsqrtf(var + 1e-5f);
  __bf16* lo = ln_out + (size_t)row * DM + d0;
#pragma unroll
  for (int i = 0; i < 4; i++)
    lo[i] = f2bf((x[i] - mu) * rs * lng[d0 + i] + lnb[d0 + i]);
}

// ---------------------------------------------------------------------------
// WMMA GEMM: C[M,N] = A[M,K] (bf16, row-major) x W[N,K]^T (bf16, row-major).
// Block = 256 thr = 8 waves; block owns a 16-row stripe; waves stride 16-col
// tiles. Fragment layout per CDNA5 ISA (16-bit A 16x32 / B 32x16): lane L<16
// holds row/col L with K = kb..kb+7 and kb+16..kb+23; lane L+16 the other
// halves. Both operands are K-contiguous -> two 16B loads each, no transpose.
// ---------------------------------------------------------------------------
#define GM_F32       0   // of = val (+bias)
#define GM_F32_ACC   1   // of += scale * (val + bias)
#define GM_BF16      2   // ob = bf16(val)
#define GM_GELU_BF16 3   // ob = bf16(gelu(val + bias))   (tanh approx, as jax.nn.gelu)

__global__ void gemm_wmma_kernel(const __bf16* __restrict__ A,
                                 const __bf16* __restrict__ W,
                                 const float* __restrict__ bias,
                                 float* __restrict__ of,
                                 __bf16* __restrict__ ob,
                                 int M, int N, int K, int mode,
                                 const float* __restrict__ sp, int smode) {
  int m0 = blockIdx.x * 16;
  int wave = threadIdx.x >> 5;
  int lane = threadIdx.x & 31;
  int r16 = lane & 15;
  int half = lane >> 4;
  const __bf16* arow = A + (size_t)(m0 + r16) * K;

  float scale = 1.0f;
  if (smode == 1) scale = sigm(sp[0]);
  else if (smode == 2) scale = 1.0f - sigm(sp[0]);

  for (int nt = wave; nt < (N >> 4); nt += 8) {
    const __bf16* wrow = W + (size_t)(nt * 16 + r16) * K;
    v8f c = {};
    for (int kb = 0; kb < K; kb += 32) {
      v8bf a0 = *(const v8bf*)(arow + kb + half * 8);
      v8bf a1 = *(const v8bf*)(arow + kb + 16 + half * 8);
      v8bf b0 = *(const v8bf*)(wrow + kb + half * 8);
      v8bf b1 = *(const v8bf*)(wrow + kb + 16 + half * 8);
      v16bf af = __builtin_shufflevector(a0, a1, 0, 1, 2, 3, 4, 5, 6, 7,
                                         8, 9, 10, 11, 12, 13, 14, 15);
      v16bf bf = __builtin_shufflevector(b0, b1, 0, 1, 2, 3, 4, 5, 6, 7,
                                         8, 9, 10, 11, 12, 13, 14, 15);
      c = __builtin_amdgcn_wmma_f32_16x16x32_bf16(false, af, false, bf,
                                                  (short)0, c, false, false);
    }
    // C/D layout: lane(0-15)=col N, VGPR r -> row r (lanes 0-15) / r+8 (16-31)
    int col = nt * 16 + r16;
    float bv = bias ? bias[col] : 0.0f;
#pragma unroll
    for (int rr = 0; rr < 8; rr++) {
      int row = m0 + rr + half * 8;
      size_t oi = (size_t)row * N + col;
      float val = c[rr] + bv;
      if (mode == GM_F32) {
        of[oi] = val;
      } else if (mode == GM_F32_ACC) {
        of[oi] += scale * val;
      } else if (mode == GM_BF16) {
        ob[oi] = f2bf(val);
      } else {  // GM_GELU_BF16
        float t = 0.7978845608028654f * (val + 0.044715f * val * val * val);
        ob[oi] = f2bf(0.5f * val * (1.0f + tanhf(t)));
      }
    }
  }
}

// ---------------------------------------------------------------------------
// Edge pass 1: logits + segment max. One wave per edge, 4 lanes per head.
// ---------------------------------------------------------------------------
__global__ void edge_logits_kernel(const int* __restrict__ eu, const int* __restrict__ ev,
                                   const float* __restrict__ dt,
                                   const float* __restrict__ Q, const float* __restrict__ Kf,
                                   const float* __restrict__ lam_p,
                                   const float* __restrict__ logtau,
                                   float* __restrict__ logits,
                                   unsigned* __restrict__ segmax) {
  int e = blockIdx.x * (blockDim.x >> 5) + (threadIdx.x >> 5);
  if (e >= NEDGE) return;
  int lane = threadIdx.x & 31;
  int hh = lane >> 2;           // 8 heads x 4 lanes
  int part = lane & 3;          // 4 floats each
  int u = eu[e], v = ev[e];
  const float4 q = *(const float4*)(Q + (size_t)v * DM + hh * DHD + part * 4);
  const float4 k = *(const float4*)(Kf + (size_t)u * DM + hh * DHD + part * 4);
  float s = q.x * k.x + q.y * k.y + q.z * k.z + q.w * k.w;
  s += __shfl_xor(s, 1, 32);
  s += __shfl_xor(s, 2, 32);
  if (part == 0) {
    float lam = fmaxf(lam_p[0], 0.0f);
    float tau = fminf(fmaxf(__expf(logtau[hh]), 0.5f), 2.0f);
    float lg = (s * 0.25f /* 1/sqrt(16) */ - lam * fmaxf(dt[e], 0.0f)) / tau;
    logits[(size_t)e * NH + hh] = lg;
    atomicMax(&segmax[(size_t)v * NH + hh], fkey(lg));
  }
}

// ---------------------------------------------------------------------------
// Edge pass 2: ex = exp(logit - segmax); denom += ex; acc_exv += ex*V[u];
// acc_v += V[u].  One wave per edge; each lane owns 4 channels.
// ---------------------------------------------------------------------------
__global__ void edge_accum_kernel(const int* __restrict__ eu, const int* __restrict__ ev,
                                  const float* __restrict__ Vf,
                                  const float* __restrict__ logits,
                                  const unsigned* __restrict__ segmax,
                                  float* __restrict__ denom,
                                  float* __restrict__ acc_exv,
                                  float* __restrict__ acc_v) {
  int e = blockIdx.x * (blockDim.x >> 5) + (threadIdx.x >> 5);
  if (e >= NEDGE) return;
  int lane = threadIdx.x & 31;
  int hh = lane >> 2;
  int u = eu[e], v = ev[e];
  float m = funkey(segmax[(size_t)v * NH + hh]);
  float ex = __expf(logits[(size_t)e * NH + hh] - m);
  if ((lane & 3) == 0) atomicAdd(&denom[(size_t)v * NH + hh], ex);
  int d = lane * 4;
  const float4 vv = *(const float4*)(Vf + (size_t)u * DM + d);
  float* pe = acc_exv + (size_t)v * DM + d;
  float* pv = acc_v + (size_t)v * DM + d;
  atomicAdd(pe + 0, ex * vv.x); atomicAdd(pe + 1, ex * vv.y);
  atomicAdd(pe + 2, ex * vv.z); atomicAdd(pe + 3, ex * vv.w);
  atomicAdd(pv + 0, vv.x); atomicAdd(pv + 1, vv.y);
  atomicAdd(pv + 2, vv.z); atomicAdd(pv + 3, vv.w);
}

// ---------------------------------------------------------------------------
// Hmix = sigmoid(mix)*acc_exv/denom + (1-sigmoid)*acc_v/max(deg,1) -> bf16
// ---------------------------------------------------------------------------
__global__ void mix_kernel(const float* __restrict__ acc_exv,
                           const float* __restrict__ acc_v,
                           const float* __restrict__ denom,
                           const int* __restrict__ deg_in,
                           const float* __restrict__ mix_p,
                           __bf16* __restrict__ hmix) {
  int idx = blockIdx.x * blockDim.x + threadIdx.x;
  if (idx >= NDST * DM) return;
  int v = idx >> 7;
  int d = idx & 127;
  int hh = d >> 4;
  float den = denom[(size_t)v * NH + hh] + 1e-16f;
  float attn = acc_exv[idx] / den;
  float dg = fmaxf((float)deg_in[v], 1.0f);
  float mean = acc_v[idx] / dg;
  float mm = sigm(mix_p[0]);
  hmix[idx] = f2bf(mm * attn + (1.0f - mm) * mean);
}

// ---------------------------------------------------------------------------
// Host launcher
// ---------------------------------------------------------------------------
static inline char* take(char*& p, size_t bytes) {
  char* r = p;
  p += (bytes + 255) & ~(size_t)255;
  return r;
}

extern "C" void kernel_launch(void* const* d_in, const int* in_sizes, int n_in,
                              void* d_out, int out_size, void* d_ws, size_t ws_size,
                              hipStream_t stream) {
  (void)in_sizes; (void)n_in; (void)out_size; (void)ws_size;
  const float* h_src   = (const float*)d_in[0];
  const float* h_dst   = (const float*)d_in[1];
  const float* x_dst   = (const float*)d_in[2];
  const int*   eidx    = (const int*)d_in[3];
  const float* edt     = (const float*)d_in[4];
  const float* W_Q     = (const float*)d_in[5];
  const float* W_K     = (const float*)d_in[6];
  const float* W_V     = (const float*)d_in[7];
  const float* W_O     = (const float*)d_in[8];
  const float* log_tau = (const float*)d_in[9];
  const float* emb_in  = (const float*)d_in[10];
  const float* emb_out = (const float*)d_in[11];
  const float* alpha_c = (const float*)d_in[12];
  const float* lam_dt  = (const float*)d_in[13];
  const float* mix_at  = (const float*)d_in[14];
  const float* gate    = (const float*)d_in[15];
  const float* ln_q_g  = (const float*)d_in[16];
  const float* ln_q_b  = (const float*)d_in[17];
  const float* ln_kv_g = (const float*)d_in[18];
  const float* ln_kv_b = (const float*)d_in[19];
  const float* fln_g   = (const float*)d_in[20];
  const float* fln_b   = (const float*)d_in[21];
  const float* ffn_w1  = (const float*)d_in[22];
  const float* ffn_b1  = (const float*)d_in[23];
  const float* ffn_w2  = (const float*)d_in[24];
  const float* ffn_b2  = (const float*)d_in[25];
  const float* wl_w    = (const float*)d_in[26];
  const float* wl_b    = (const float*)d_in[27];
  const float* w0_w    = (const float*)d_in[28];
  const float* w0_b    = (const float*)d_in[29];
  const float* w1_w    = (const float*)d_in[30];
  const float* w1_b    = (const float*)d_in[31];

  const int* eu = eidx;
  const int* ev = eidx + NEDGE;
  float* out_f = (float*)d_out;

  // ---- workspace layout (zero-init block first, one memset) ----
  char* p = (char*)d_ws;
  char* zbase   = p;
  int*   deg_out = (int*)  take(p, (size_t)NSRC * 4);
  int*   deg_in  = (int*)  take(p, (size_t)NDST * 4);
  float* denom   = (float*)take(p, (size_t)NDST * NH * 4);
  float* acc_exv = (float*)take(p, (size_t)NDST * DM * 4);
  float* acc_v   = (float*)take(p, (size_t)NDST * DM * 4);
  size_t zbytes = (size_t)(p - zbase);

  unsigned* segmax = (unsigned*)take(p, (size_t)NDST * NH * 4);
  float* logits = (float*)take(p, (size_t)NEDGE * NH * 4);
  float* Qf = (float*)take(p, (size_t)NDST * DM * 4);
  float* Kf = (float*)take(p, (size_t)NSRC * DM * 4);
  float* Vf = (float*)take(p, (size_t)NSRC * DM * 4);
  __bf16* Hq_bf   = (__bf16*)take(p, (size_t)NDST * DM * 2);
  __bf16* Hk_bf   = (__bf16*)take(p, (size_t)NSRC * DM * 2);
  __bf16* hdst_bf = (__bf16*)take(p, (size_t)NDST * DM * 2);
  __bf16* xdst_bf = (__bf16*)take(p, (size_t)NDST * DM * 2);
  __bf16* hmix_bf = (__bf16*)take(p, (size_t)NDST * DM * 2);
  __bf16* hout_bf = (__bf16*)take(p, (size_t)NDST * DM * 2);
  __bf16* z_bf    = (__bf16*)take(p, (size_t)NDST * DM * 2);
  __bf16* hid_bf  = (__bf16*)take(p, (size_t)NDST * 4 * DM * 2);
  __bf16* wq_bf = (__bf16*)take(p, (size_t)DM * DM * 2);
  __bf16* wk_bf = (__bf16*)take(p, (size_t)DM * DM * 2);
  __bf16* wv_bf = (__bf16*)take(p, (size_t)DM * DM * 2);
  __bf16* wo_bf = (__bf16*)take(p, (size_t)DM * DM * 2);
  __bf16* wl_bf = (__bf16*)take(p, (size_t)DM * DM * 2);
  __bf16* w0_bf = (__bf16*)take(p, (size_t)DM * DM * 2);
  __bf16* w1_bf = (__bf16*)take(p, (size_t)DM * DM * 2);
  __bf16* fw1_bf = (__bf16*)take(p, (size_t)DM * 4 * DM * 2);  // [512,128]
  __bf16* fw2_bf = (__bf16*)take(p, (size_t)4 * DM * DM * 2);  // [128,512]

  // ---- init ----
  hipMemsetAsync(zbase, 0, zbytes, stream);
  segmax_init_kernel<<<(NDST * NH + 255) / 256, 256, 0, stream>>>(segmax, NDST * NH);

  // ---- degrees ----
  deg_kernel<<<(NEDGE + 255) / 256, 256, 0, stream>>>(eu, ev, deg_out, deg_in);

  // ---- centrality + LN (bf16 activations) ----
  int rows_blk = (NSRC + 7) / 8;  // 8 wave32 rows per 256-thread block
  cent_ln_kernel<<<rows_blk, 256, 0, stream>>>(h_src, deg_out, 1, emb_in, emb_out,
                                               alpha_c, ln_kv_g, ln_kv_b,
                                               Hk_bf, nullptr, NSRC);
  cent_ln_kernel<<<rows_blk, 256, 0, stream>>>(h_dst, deg_in, 0, emb_in, emb_out,
                                               alpha_c, ln_q_g, ln_q_b,
                                               Hq_bf, hdst_bf, NDST);
  f2bf_kernel<<<(NDST * DM + 255) / 256, 256, 0, stream>>>(x_dst, xdst_bf, NDST * DM);

  // ---- weights -> bf16 (ffn weights transposed to [Nout, K]) ----
  f2bf_kernel<<<(DM * DM + 255) / 256, 256, 0, stream>>>(W_Q, wq_bf, DM * DM);
  f2bf_kernel<<<(DM * DM + 255) / 256, 256, 0, stream>>>(W_K, wk_bf, DM * DM);
  f2bf_kernel<<<(DM * DM + 255) / 256, 256, 0, stream>>>(W_V, wv_bf, DM * DM);
  f2bf_kernel<<<(DM * DM + 255) / 256, 256, 0, stream>>>(W_O, wo_bf, DM * DM);
  f2bf_kernel<<<(DM * DM + 255) / 256, 256, 0, stream>>>(wl_w, wl_bf, DM * DM);
  f2bf_kernel<<<(DM * DM + 255) / 256, 256, 0, stream>>>(w0_w, w0_bf, DM * DM);
  f2bf_kernel<<<(DM * DM + 255) / 256, 256, 0, stream>>>(w1_w, w1_bf, DM * DM);
  f2bfT_kernel<<<(DM * 4 * DM + 255) / 256, 256, 0, stream>>>(ffn_w1, fw1_bf, DM, 4 * DM);
  f2bfT_kernel<<<(DM * 4 * DM + 255) / 256, 256, 0, stream>>>(ffn_w2, fw2_bf, 4 * DM, DM);

  // ---- Q/K/V projections (WMMA) ----
  gemm_wmma_kernel<<<NDST / 16, 256, 0, stream>>>(Hq_bf, wq_bf, nullptr, Qf, nullptr,
                                                  NDST, DM, DM, GM_F32, nullptr, 0);
  gemm_wmma_kernel<<<NSRC / 16, 256, 0, stream>>>(Hk_bf, wk_bf, nullptr, Kf, nullptr,
                                                  NSRC, DM, DM, GM_F32, nullptr, 0);
  gemm_wmma_kernel<<<NSRC / 16, 256, 0, stream>>>(Hk_bf, wv_bf, nullptr, Vf, nullptr,
                                                  NSRC, DM, DM, GM_F32, nullptr, 0);

  // ---- edge attention ----
  int eblk = (NEDGE + 7) / 8;  // 1 wave per edge
  edge_logits_kernel<<<eblk, 256, 0, stream>>>(eu, ev, edt, Qf, Kf, lam_dt, log_tau,
                                               logits, segmax);
  edge_accum_kernel<<<eblk, 256, 0, stream>>>(eu, ev, Vf, logits, segmax,
                                              denom, acc_exv, acc_v);
  mix_kernel<<<(NDST * DM + 255) / 256, 256, 0, stream>>>(acc_exv, acc_v, denom,
                                                          deg_in, mix_at, hmix_bf);

  // ---- output projection + gated mixing ----
  gemm_wmma_kernel<<<NDST / 16, 256, 0, stream>>>(hmix_bf, wo_bf, nullptr, nullptr,
                                                  hout_bf, NDST, DM, DM, GM_BF16,
                                                  nullptr, 0);
  gemm_wmma_kernel<<<NDST / 16, 256, 0, stream>>>(hout_bf, wl_bf, wl_b, out_f, nullptr,
                                                  NDST, DM, DM, GM_F32, nullptr, 0);
  gemm_wmma_kernel<<<NDST / 16, 256, 0, stream>>>(hdst_bf, w0_bf, w0_b, out_f, nullptr,
                                                  NDST, DM, DM, GM_F32_ACC, gate, 2);
  gemm_wmma_kernel<<<NDST / 16, 256, 0, stream>>>(xdst_bf, w1_bf, w1_b, out_f, nullptr,
                                                  NDST, DM, DM, GM_F32_ACC, gate, 1);

  // ---- FFN: out += gelu(LN(out) @ w1 + b1) @ w2 + b2 ----
  cent_ln_kernel<<<(NDST + 7) / 8, 256, 0, stream>>>(out_f, nullptr, 0, nullptr, nullptr,
                                                     nullptr, fln_g, fln_b,
                                                     z_bf, nullptr, NDST);
  gemm_wmma_kernel<<<NDST / 16, 256, 0, stream>>>(z_bf, fw1_bf, ffn_b1, nullptr, hid_bf,
                                                  NDST, 4 * DM, DM, GM_GELU_BF16,
                                                  nullptr, 0);
  gemm_wmma_kernel<<<NDST / 16, 256, 0, stream>>>(hid_bf, fw2_bf, ffn_b2, out_f, nullptr,
                                                  NDST, DM, 4 * DM, GM_F32_ACC,
                                                  nullptr, 0);
}
